// GraphConv_89154931130782
// MI455X (gfx1250) — matrix operand order repaired and checked by reading, not compiled, not run on hardware
//
#include <hip/hip_runtime.h>
#include <stdint.h>

#define NU 8000
#define NI 20000
#define NT 4000
#define NN (NU + NI + NT)
#define DD 64
#define NE 500000
#define LDSTRIDE 68          // 64 + 4 pad: conflict-free column reads
#define ENC_NEG_INF 0x007FFFFFu

typedef __attribute__((ext_vector_type(16))) __bf16 v16bf;
typedef __attribute__((ext_vector_type(8)))  float  v8f;

struct Nodes {
    const float* uc; const float* ic; const float* tc;   // centers
    const float* uo; const float* io; const float* tg;   // offsets
};

// ---- monotone float<->uint mapping for atomic max over signed floats ----
__device__ __forceinline__ unsigned enc_f32(float f) {
    unsigned u = __float_as_uint(f);
    return (u & 0x80000000u) ? ~u : (u | 0x80000000u);
}
__device__ __forceinline__ float dec_f32(unsigned u) {
    unsigned b = (u & 0x80000000u) ? (u ^ 0x80000000u) : ~u;
    return __uint_as_float(b);
}

__device__ __forceinline__ const float* center_row(const Nodes& nd, int n) {
    if (n < NU) return nd.uc + (size_t)n * DD;
    if (n < NU + NI) return nd.ic + (size_t)(n - NU) * DD;
    return nd.tc + (size_t)(n - NU - NI) * DD;
}
__device__ __forceinline__ float offset_val(const Nodes& nd, int n, int d) {
    float v;
    if (n < NU) v = nd.uo[(size_t)n * DD + d];
    else if (n < NU + NI) v = nd.io[(size_t)(n - NU) * DD + d];
    else v = nd.tg[(size_t)(n - NU - NI) * DD + d];
    return v > 0.f ? v : 0.f;
}

// ---------------- init: accumulators + output ----------------
__global__ void init_k(unsigned* segmax, float* den, float* num,
                       unsigned* count, float* out) {
    int i = blockIdx.x * blockDim.x + threadIdx.x;
    const int W = (NU + 1) * DD;
    if (i < W) { segmax[i] = ENC_NEG_INF; den[i] = 0.f; num[i] = 0.f; }
    if (i < NU * DD) { out[i] = 0.f; out[NU * DD + i] = 0.f; }
    if (i == 0) *count = 0u;
}

// ---------------- compact edges with head < NU ----------------
__global__ void compact_k(const int* head, unsigned* count, unsigned* list) {
    int e = blockIdx.x * blockDim.x + threadIdx.x;
    bool q = (e < NE) && (head[e] < NU);
#if __has_builtin(__builtin_amdgcn_ballot_w32)
    unsigned mask = __builtin_amdgcn_ballot_w32(q);
#else
    unsigned mask = (unsigned)__ballot(q);
#endif
    int lane = threadIdx.x & 31;
    unsigned base = 0;
    int tot = __popc(mask);
    if (lane == 0 && tot) base = atomicAdd(count, (unsigned)tot);
    base = __shfl(base, 0, 32);
    if (q) list[base + __popc(mask & ((1u << lane) - 1u))] = (unsigned)e;
}

// ---------------- WMMA center-net passes ----------------
// PASS 1: logits -> per-segment max (encoded atomic max)
// PASS 2: exp(logit - max) -> atomic-add denominator and numerator(*src)
template <int PASS>
__global__ __launch_bounds__(128) void mlp_pass(
    Nodes nd, const int* head, const int* tail,
    const unsigned* count_p, const unsigned* list,
    const float* W1, const float* B1, const float* W2, const float* B2,
    unsigned* segmax, float* den, float* num)
{
    __shared__ float hbuf[4][16 * LDSTRIDE];
    __shared__ int   tls[4][16];
    __shared__ int   hds[4][16];

    const int wave = threadIdx.x >> 5;
    const int lane = threadIdx.x & 31;
    const unsigned count = *count_p;
    const unsigned tile = blockIdx.x * 4u + (unsigned)wave;
    if (tile * 16u >= count) return;                    // wave-uniform exit

    if (lane < 16) {                                    // tile metadata
        unsigned slot = tile * 16u + (unsigned)lane;
        int h = NU, t = 0;
        if (slot < count) { unsigned e = list[slot]; h = head[e]; t = tail[e]; }
        hds[wave][lane] = h;
        tls[wave][lane] = t;
    }
    asm volatile("s_wait_dscnt 0x0" ::: "memory");

    const int Mrow = lane & 15;
    const int grp  = lane >> 4;
    const int Ncol = lane & 15;
    const int tailM = tls[wave][Mrow];
    const float* srow = center_row(nd, tailM);

    // A fragments (16x32 bf16, ISA 16-bit A layout), K-blocks 0/1
    v16bf a0, a1;
#pragma unroll
    for (int j = 0; j < 8; ++j) {
        int k = grp * 8 + ((j < 4) ? 2 * j : 16 + 2 * (j - 4));
        a0[2 * j]     = (__bf16)srow[k];
        a0[2 * j + 1] = (__bf16)srow[k + 1];
        a1[2 * j]     = (__bf16)srow[32 + k];
        a1[2 * j + 1] = (__bf16)srow[32 + k + 1];
    }

    // ---- layer 1: h = relu(src @ W1^T + b1) -> LDS ----
#pragma unroll
    for (int n = 0; n < 4; ++n) {
        v8f acc = {0.f, 0.f, 0.f, 0.f, 0.f, 0.f, 0.f, 0.f};
#pragma unroll
        for (int kb = 0; kb < 2; ++kb) {
            v16bf bf;
            const float* wrow = W1 + (size_t)(n * 16 + Ncol) * DD + kb * 32 + grp * 8;
#pragma unroll
            for (int v = 0; v < 8; ++v) {
                bf[2 * v]     = (__bf16)wrow[(v < 4) ? 2 * v : 16 + 2 * (v - 4)];
                bf[2 * v + 1] = (__bf16)wrow[((v < 4) ? 2 * v : 16 + 2 * (v - 4)) + 1];
            }
            acc = __builtin_amdgcn_wmma_f32_16x16x32_bf16(
                false, (kb == 0) ? a0 : a1, false, bf, (short)0, acc, false, false);
        }
        float bias = B1[n * 16 + Ncol];
        int Mb = grp * 8;
#pragma unroll
        for (int r = 0; r < 8; ++r) {
            float hv = acc[r] + bias;
            hbuf[wave][(Mb + r) * LDSTRIDE + n * 16 + Ncol] = hv > 0.f ? hv : 0.f;
        }
    }
    asm volatile("s_wait_dscnt 0x0" ::: "memory");

    // A2 fragments from LDS h
    const float* hrow = &hbuf[wave][Mrow * LDSTRIDE];
    v16bf c0, c1;
#pragma unroll
    for (int j = 0; j < 8; ++j) {
        int k = grp * 8 + ((j < 4) ? 2 * j : 16 + 2 * (j - 4));
        c0[2 * j]     = (__bf16)hrow[k];
        c0[2 * j + 1] = (__bf16)hrow[k + 1];
        c1[2 * j]     = (__bf16)hrow[32 + k];
        c1[2 * j + 1] = (__bf16)hrow[32 + k + 1];
    }

    // ---- layer 2: logits = h @ W2^T + b2 ----
#pragma unroll
    for (int n = 0; n < 4; ++n) {
        v8f acc = {0.f, 0.f, 0.f, 0.f, 0.f, 0.f, 0.f, 0.f};
#pragma unroll
        for (int kb = 0; kb < 2; ++kb) {
            v16bf bf;
            const float* wrow = W2 + (size_t)(n * 16 + Ncol) * DD + kb * 32 + grp * 8;
#pragma unroll
            for (int v = 0; v < 8; ++v) {
                bf[2 * v]     = (__bf16)wrow[(v < 4) ? 2 * v : 16 + 2 * (v - 4)];
                bf[2 * v + 1] = (__bf16)wrow[((v < 4) ? 2 * v : 16 + 2 * (v - 4)) + 1];
            }
            acc = __builtin_amdgcn_wmma_f32_16x16x32_bf16(
                false, (kb == 0) ? c0 : c1, false, bf, (short)0, acc, false, false);
        }
        float bias = B2[n * 16 + Ncol];
        int Mb = grp * 8;
#pragma unroll
        for (int r = 0; r < 8; ++r) {
            float logit = acc[r] + bias;
            int M = Mb + r;
            int hseg = hds[wave][M];
            unsigned off = (unsigned)hseg * 64u + (unsigned)(n * 16 + Ncol);
            if (PASS == 1) {
                atomicMax(&segmax[off], enc_f32(logit));
            } else {
                float mv = dec_f32(segmax[off]);
                float ev = __expf(logit - mv);
                atomicAdd(&den[off], ev);
                const float* sr = center_row(nd, tls[wave][M]);
                atomicAdd(&num[off], ev * sr[n * 16 + Ncol]);
            }
        }
    }
}

// ---------------- offset scatter-max (union of all 5 masks) ----------------
__global__ void offmax_k(Nodes nd, const int* head, const int* tail,
                         int require_tail_ge_nu, unsigned* outbits) {
    long long g = (long long)blockIdx.x * blockDim.x + threadIdx.x;
    int e = (int)(g >> 6);
    int d = (int)(g & 63);
    if (e >= NE) return;
    int h = head[e];
    if (h >= NU) return;
    int t = tail[e];
    if (require_tail_ge_nu && t < NU) return;
    float v = offset_val(nd, t, d);      // v >= 0 -> uint bits are order-preserving
    atomicMax(&outbits[(unsigned)h * 64u + (unsigned)d], __float_as_uint(v));
}

// ---------------- finalize: emb = l2norm(num/den) ----------------
__global__ __launch_bounds__(256) void finalize_k(const float* num, const float* den,
                                                  float* out_emb) {
    int wave = threadIdx.x >> 5, lane = threadIdx.x & 31;
    int u = blockIdx.x * 8 + wave;
    if (u >= NU) return;
    size_t b = (size_t)u * 64;
    float d0 = den[b + lane],      n0 = num[b + lane];
    float d1 = den[b + lane + 32], n1 = num[b + lane + 32];
    float e0 = d0 > 0.f ? n0 / d0 : 0.f;
    float e1 = d1 > 0.f ? n1 / d1 : 0.f;
    float ss = e0 * e0 + e1 * e1;
#pragma unroll
    for (int w = 16; w >= 1; w >>= 1) ss += __shfl_xor(ss, w, 32);
    float nrm = sqrtf(ss);
    nrm = nrm > 1e-12f ? nrm : 1e-12f;
    out_emb[b + lane]      = e0 / nrm;
    out_emb[b + lane + 32] = e1 / nrm;
}

extern "C" void kernel_launch(void* const* d_in, const int* in_sizes, int n_in,
                              void* d_out, int out_size, void* d_ws, size_t ws_size,
                              hipStream_t stream) {
    (void)in_sizes; (void)n_in; (void)out_size; (void)ws_size;
    const float* user_center = (const float*)d_in[0];
    const float* user_offset = (const float*)d_in[1];
    const float* item_center = (const float*)d_in[2];
    const float* item_offset = (const float*)d_in[3];
    const float* tag_center  = (const float*)d_in[4];
    const float* tag_offset  = (const float*)d_in[5];
    const int*   head1 = (const int*)d_in[6];
    const int*   tail1 = (const int*)d_in[7];
    const int*   head2 = (const int*)d_in[8];
    const int*   tail2 = (const int*)d_in[9];
    // d_in[10] visit_time, d_in[15..18] tW1/tb1/tW2/tb2: dead (lam == 1.0)
    const float* cW1 = (const float*)d_in[11];
    const float* cb1 = (const float*)d_in[12];
    const float* cW2 = (const float*)d_in[13];
    const float* cb2 = (const float*)d_in[14];

    char* ws = (char*)d_ws;
    const size_t seg_bytes = (size_t)(NU + 1) * DD * 4;     // ~2.05 MB
    unsigned* count  = (unsigned*)ws;
    unsigned* segmax = (unsigned*)(ws + 256);
    float*    den    = (float*)(ws + 256 + seg_bytes);
    float*    num    = (float*)(ws + 256 + 2 * seg_bytes);
    unsigned* list   = (unsigned*)(ws + 256 + 3 * seg_bytes); // +2 MB => ~8.2 MB total

    float* out = (float*)d_out;                    // [0, NU*64): emb, [NU*64, 2*NU*64): offset
    Nodes nd{user_center, item_center, tag_center, user_offset, item_offset, tag_offset};

    init_k<<<((NU + 1) * DD + 255) / 256, 256, 0, stream>>>(segmax, den, num, count, out);
    compact_k<<<(NE + 255) / 256, 256, 0, stream>>>(head1, count, list);

    const int mlp_blocks = (NE / 16 + 3) / 4;      // 4 waves/block, 16 edges/wave
    mlp_pass<1><<<mlp_blocks, 128, 0, stream>>>(nd, head1, tail1, count, list,
                                                cW1, cb1, cW2, cb2, segmax, den, num);
    mlp_pass<2><<<mlp_blocks, 128, 0, stream>>>(nd, head1, tail1, count, list,
                                                cW1, cb1, cW2, cb2, segmax, den, num);

    unsigned* offbits = (unsigned*)out + (size_t)NU * DD;
    const int off_blocks = (int)(((long long)NE * 64 + 255) / 256);
    offmax_k<<<off_blocks, 256, 0, stream>>>(nd, head1, tail1, 1, offbits);
    offmax_k<<<off_blocks, 256, 0, stream>>>(nd, head2, tail2, 0, offbits);

    finalize_k<<<(NU + 7) / 8, 256, 0, stream>>>(num, den, out);
}
